// AttentionBlock_22462678958608
// MI455X (gfx1250) — matrix operand order repaired and checked by reading, not compiled, not run on hardware
//
#include <hip/hip_runtime.h>
#include <hip/hip_bf16.h>
#include <math.h>

// ---------------------------------------------------------------------------
// AttentionBlock (guided-diffusion style) for MI455X / gfx1250, wave32 WMMA.
// Shapes: B=16, C=512, T=1024, HEADS=8, HC=64, L=77, CROSS=768.
// All GEMMs use v_wmma_f32_16x16x32_f16 (f16 in, f32 accumulate).
// Attention is flash-style (streaming softmax), logits never materialized.
// Register blocking: GEMM waves compute 32x64 tiles (8 WMMAs / k-step),
// attention waves own 32 query rows (every K/V fragment feeds 2 WMMAs).
// ---------------------------------------------------------------------------

typedef _Float16 half_t;
typedef _Float16 v16h __attribute__((ext_vector_type(16)));
typedef _Float16 v8h  __attribute__((ext_vector_type(8)));
typedef float    v8f  __attribute__((ext_vector_type(8)));

#define B_      16
#define C_      512
#define T_      1024
#define HEADS_  8
#define HC_     64
#define L_      77
#define CROSS_  768
#define S_TOT   1101          /* T_ + L_ */
#define S_PAD   1120          /* padded to multiple of 32 */
#define NEG_BIG (-1e30f)

static __device__ __forceinline__ v8f wmma16(v16h a, v16h b, v8f c) {
  return __builtin_amdgcn_wmma_f32_16x16x32_f16(false, a, false, b,
                                                (short)0, c, false, false);
}

// A-matrix fragment (16x32 f16, M x K). Lane holds row m = lane&15.
// halves 0..15 = A[m][2*kb8 .. 2*kb8+15] (contiguous),
// kb8 = 8*(lane>=16).  `row` points at A[m][k0].
static __device__ __forceinline__ v16h load_a_frag(const half_t* row, int kb8) {
  v8h lo = *(const v8h*)(row + 2 * kb8);
  v8h hi = *(const v8h*)(row + 2 * kb8 + 8);
  v16h r;
#pragma unroll
  for (int i = 0; i < 8; ++i) { r[i] = lo[i]; r[8 + i] = hi[i]; }
  return r;
}

// ---------------------------------------------------------------------------
// K0: convert weights f32 -> f16 into workspace
// ---------------------------------------------------------------------------
__global__ __launch_bounds__(256) void cvt_kernel(
    const float* __restrict__ wq, const float* __restrict__ wc,
    const float* __restrict__ wp, half_t* __restrict__ wqh,
    half_t* __restrict__ wch, half_t* __restrict__ wph) {
  int i = blockIdx.x * 256 + threadIdx.x;
  if (i < 3 * C_ * C_)     wqh[i] = (half_t)wq[i];
  if (i < 2 * C_ * CROSS_) wch[i] = (half_t)wc[i];
  if (i < C_ * C_)         wph[i] = (half_t)wp[i];
}

// ---------------------------------------------------------------------------
// K1: GroupNorm(32) + affine; writes xn transposed as f16 [B, T, C]
// ---------------------------------------------------------------------------
__global__ __launch_bounds__(256) void gn_kernel(
    const float* __restrict__ x, const float* __restrict__ gamma,
    const float* __restrict__ beta, half_t* __restrict__ xn_t) {
  int bg = blockIdx.x;
  int g = bg & 31, b = bg >> 5;
  int ch0 = g * 16;
  const float* xb = x + ((size_t)b * C_ + ch0) * T_;
  float s = 0.f, ss = 0.f;
  for (int e = threadIdx.x; e < 16 * T_; e += 256) {
    float v = xb[e];
    s += v; ss += v * v;
  }
  __shared__ float rs[256], rq[256];
  int tid = threadIdx.x;
  rs[tid] = s; rq[tid] = ss;
  __syncthreads();
  for (int w = 128; w > 0; w >>= 1) {
    if (tid < w) { rs[tid] += rs[tid + w]; rq[tid] += rq[tid + w]; }
    __syncthreads();
  }
  float mean = rs[0] * (1.f / 16384.f);
  float var  = rq[0] * (1.f / 16384.f) - mean * mean;
  float inv  = rsqrtf(var + 1e-5f);
  for (int e = tid; e < 16 * T_; e += 256) {
    int ch = ch0 + (e >> 10);
    int t  = e & (T_ - 1);
    float v = (xb[e] - mean) * inv * gamma[ch] + beta[ch];
    xn_t[((size_t)b * T_ + t) * C_ + ch] = (half_t)v;
  }
}

// ---------------------------------------------------------------------------
// K2: qkv = W_qkv[1536,512] @ xn + b ; one wave per 32x64 output tile.
//   q -> qT [B,H,T,64]      (t-major: attention A-fragments contiguous)
//   k -> kT [B,H,S_PAD,64]  (s-major: logits B-fragments contiguous)
//   v -> vx [B,H,64,S_PAD]  (c-major: P*V B-fragments contiguous)
// ---------------------------------------------------------------------------
__global__ __launch_bounds__(32) void qkv_kernel(
    const half_t* __restrict__ wq, const float* __restrict__ bias,
    const half_t* __restrict__ xn_t, half_t* __restrict__ qT,
    half_t* __restrict__ kT, half_t* __restrict__ vx) {
  int idx = blockIdx.x;
  int nb = idx & 15; idx >>= 4;     // 16 n-blocks of 64 columns
  int mb = idx % 48;                // 48 m-blocks of 32 rows
  int b  = idx / 48;
  int o0 = mb * 32, t0 = nb * 64;
  int lane = threadIdx.x;
  int ncol = lane & 15;
  int kb8  = (lane >> 4) * 8;
  int kb16 = (lane >> 4) * 16;
  const half_t* arow0 = wq + (size_t)(o0 + ncol) * C_;
  const half_t* arow1 = wq + (size_t)(o0 + 16 + ncol) * C_;
  const half_t* bbase = xn_t + ((size_t)b * T_ + t0 + ncol) * C_;
  v8f acc[2][4] = {};
#pragma unroll 2
  for (int k0 = 0; k0 < C_; k0 += 32) {
    v16h A0 = load_a_frag(arow0 + k0, kb8);
    v16h A1 = load_a_frag(arow1 + k0, kb8);
#pragma unroll
    for (int j = 0; j < 4; ++j) {
      v16h Bf = *(const v16h*)(bbase + (size_t)j * 16 * C_ + k0 + kb16);
      acc[0][j] = wmma16(A0, Bf, acc[0][j]);
      acc[1][j] = wmma16(A1, Bf, acc[1][j]);
    }
  }
#pragma unroll
  for (int i = 0; i < 2; ++i) {
#pragma unroll
    for (int j = 0; j < 4; ++j) {
#pragma unroll
      for (int r = 0; r < 8; ++r) {
        int m  = r + ((lane >> 4) << 3);
        int oc = o0 + i * 16 + m;
        int t  = t0 + j * 16 + ncol;
        half_t hv = (half_t)(acc[i][j][r] + bias[oc]);
        if (oc < C_) {
          qT[(((size_t)(b * HEADS_ + (oc >> 6)) * T_) + t) * HC_ + (oc & 63)] = hv;
        } else if (oc < 2 * C_) {
          int o2 = oc - C_;
          kT[(((size_t)(b * HEADS_ + (o2 >> 6)) * S_PAD) + t) * HC_ + (o2 & 63)] = hv;
        } else {
          int o2 = oc - 2 * C_;
          vx[((size_t)(b * HEADS_ + (o2 >> 6)) * HC_ + (o2 & 63)) * S_PAD + t] = hv;
        }
      }
    }
  }
}

// ---------------------------------------------------------------------------
// K3: cross kv = W_c[1024,768] @ c[B,768,77] + b_c, appended at s >= 1024
// ---------------------------------------------------------------------------
__global__ __launch_bounds__(32) void ckv_kernel(
    const half_t* __restrict__ wc, const float* __restrict__ bias,
    const float* __restrict__ cin, half_t* __restrict__ kT,
    half_t* __restrict__ vx) {
  int idx = blockIdx.x;
  int nt = idx % 5; idx /= 5;
  int mo = idx & 63;
  int b  = idx >> 6;
  int o0 = mo * 16, l0 = nt * 16;
  int lane = threadIdx.x;
  int ncol = lane & 15;
  int kb8  = (lane >> 4) * 8;
  int kb16 = (lane >> 4) * 16;
  int lcol = l0 + ncol;
  int lcl  = lcol < L_ ? lcol : L_ - 1;   // clamp loads; store is masked
  const half_t* arow = wc + (size_t)(o0 + ncol) * CROSS_;
  const float*  cb   = cin + (size_t)b * CROSS_ * L_;
  v8f acc = {};
  for (int k0 = 0; k0 < CROSS_; k0 += 32) {
    v16h A = load_a_frag(arow + k0, kb8);
    v16h Bf;
#pragma unroll
    for (int i = 0; i < 16; ++i)
      Bf[i] = (half_t)cb[(size_t)(k0 + kb16 + i) * L_ + lcl];
    acc = wmma16(A, Bf, acc);
  }
  if (lcol < L_) {
    int s = T_ + lcol;
#pragma unroll
    for (int r = 0; r < 8; ++r) {
      int m  = r + ((lane >> 4) << 3);
      int oc = o0 + m;
      half_t hv = (half_t)(acc[r] + bias[oc]);
      if (oc < C_) {
        kT[(((size_t)(b * HEADS_ + (oc >> 6)) * S_PAD) + s) * HC_ + (oc & 63)] = hv;
      } else {
        int o2 = oc - C_;
        vx[((size_t)(b * HEADS_ + (o2 >> 6)) * HC_ + (o2 & 63)) * S_PAD + s] = hv;
      }
    }
  }
}

// ---------------------------------------------------------------------------
// K4: flash attention. One wave per (b, h, 32 query rows).
//   per 32-key chunk: 8 WMMAs for logits (K frags reused by 2 row-tiles),
//   streaming softmax in LDS (1 row / lane), 8 WMMAs for O += P*V^T
//   (V frags reused by 2 row-tiles). O accumulators: 64 VGPRs.
// ---------------------------------------------------------------------------
__global__ __launch_bounds__(32) void attn_kernel(
    const half_t* __restrict__ qT, const half_t* __restrict__ kT,
    const half_t* __restrict__ vx, half_t* __restrict__ aT) {
  int idx = blockIdx.x;
  int tt = idx & 31; idx >>= 5;       // T_/32 = 32 query tiles
  int h = idx & 7;
  int b = idx >> 3;
  int t0 = tt * 32;
  int lane = threadIdx.x;
  int ncol = lane & 15;
  int kb8  = (lane >> 4) * 8;
  int kb16 = (lane >> 4) * 16;
  int hr   = (lane >> 4) << 3;        // 0 or 8: row offset in C/D layout

  __shared__ float  Ls[32][33];       // padded: breaks 64-bank conflicts
  __shared__ __attribute__((aligned(32))) half_t Ps[32][40];
  __shared__ float  m_s[32], l_s[32], sc_s[32];

  // q fragments for 2 row-tiles (K-dim = 64 ch -> 2 frags each), persistent
  const half_t* qbase = qT + (((size_t)(b * HEADS_ + h) * T_) + t0) * HC_;
  v16h A0 = load_a_frag(qbase + (size_t)ncol * HC_, kb8);
  v16h A1 = load_a_frag(qbase + (size_t)ncol * HC_ + 32, kb8);
  v16h A2 = load_a_frag(qbase + (size_t)(16 + ncol) * HC_, kb8);
  v16h A3 = load_a_frag(qbase + (size_t)(16 + ncol) * HC_ + 32, kb8);

  const half_t* kb_ = kT + (size_t)(b * HEADS_ + h) * S_PAD * HC_;
  const half_t* vb_ = vx + (size_t)(b * HEADS_ + h) * HC_ * S_PAD;

  m_s[lane] = NEG_BIG;
  l_s[lane] = 0.f;
  v8f O0[4] = {};
  v8f O1[4] = {};
  __syncthreads();

  for (int s0 = 0; s0 < S_PAD; s0 += 32) {
    // ---- logits: two 16-col key tiles; each K fragment feeds both row-tiles
#pragma unroll
    for (int st = 0; st < 2; ++st) {
      int sc = s0 + st * 16 + ncol;
      const half_t* krow = kb_ + (size_t)sc * HC_;
      v16h B0 = *(const v16h*)(krow + kb16);
      v16h B1 = *(const v16h*)(krow + 32 + kb16);
      v8f acc0 = {}, acc1 = {};
      acc0 = wmma16(A0, B0, acc0);
      acc0 = wmma16(A1, B1, acc0);
      acc1 = wmma16(A2, B0, acc1);
      acc1 = wmma16(A3, B1, acc1);
      bool valid = sc < S_TOT;
#pragma unroll
      for (int r = 0; r < 8; ++r) {
        Ls[r + hr][st * 16 + ncol]      = valid ? acc0[r] * 0.125f : NEG_BIG;
        Ls[16 + r + hr][st * 16 + ncol] = valid ? acc1[r] * 0.125f : NEG_BIG;
      }
    }
    __syncthreads();
    // ---- online softmax: each lane owns one query row
    {
      int m = lane;
      float mo_ = m_s[m], mx = mo_;
#pragma unroll
      for (int j = 0; j < 32; ++j) mx = fmaxf(mx, Ls[m][j]);
      float scl = __expf(mo_ - mx);
      float sum = 0.f;
#pragma unroll
      for (int j = 0; j < 32; ++j) {
        float p = __expf(Ls[m][j] - mx);
        Ps[m][j] = (half_t)p;
        sum += p;
      }
      m_s[m]  = mx;
      l_s[m]  = l_s[m] * scl + sum;
      sc_s[m] = scl;
    }
    __syncthreads();
    // ---- rescale O, then O += P * V^T (V frags shared by both row-tiles)
    v16h Pf0 = load_a_frag(&Ps[ncol][0], kb8);
    v16h Pf1 = load_a_frag(&Ps[16 + ncol][0], kb8);
#pragma unroll
    for (int r = 0; r < 8; ++r) {
      float sA = sc_s[r + hr];
      float sB = sc_s[16 + r + hr];
      O0[0][r] *= sA; O0[1][r] *= sA; O0[2][r] *= sA; O0[3][r] *= sA;
      O1[0][r] *= sB; O1[1][r] *= sB; O1[2][r] *= sB; O1[3][r] *= sB;
    }
#pragma unroll
    for (int ct = 0; ct < 4; ++ct) {
      v16h Bv = *(const v16h*)(vb_ + (size_t)(ct * 16 + ncol) * S_PAD + s0 + kb16);
      O0[ct] = wmma16(Pf0, Bv, O0[ct]);
      O1[ct] = wmma16(Pf1, Bv, O1[ct]);
    }
    __syncthreads();
  }
  // ---- normalize and store a^T [B, T, C] (c contiguous for proj B-frags)
  float li0[8], li1[8];
#pragma unroll
  for (int r = 0; r < 8; ++r) {
    li0[r] = 1.f / l_s[r + hr];
    li1[r] = 1.f / l_s[16 + r + hr];
  }
#pragma unroll
  for (int ct = 0; ct < 4; ++ct) {
#pragma unroll
    for (int r = 0; r < 8; ++r) {
      int col = h * HC_ + ct * 16 + ncol;
      aT[((size_t)b * T_ + t0 + r + hr) * C_ + col]      = (half_t)(O0[ct][r] * li0[r]);
      aT[((size_t)b * T_ + t0 + 16 + r + hr) * C_ + col] = (half_t)(O1[ct][r] * li1[r]);
    }
  }
}

// ---------------------------------------------------------------------------
// K5: out = x + W_p @ a + b_p ; one wave per 32x64 output tile
// ---------------------------------------------------------------------------
__global__ __launch_bounds__(32) void proj_kernel(
    const half_t* __restrict__ wp, const float* __restrict__ bias,
    const half_t* __restrict__ aT, const float* __restrict__ x,
    float* __restrict__ out) {
  int idx = blockIdx.x;
  int nb = idx & 15; idx >>= 4;     // 16 n-blocks of 64
  int mb = idx & 15;                // 16 m-blocks of 32
  int b  = idx >> 4;
  int o0 = mb * 32, t0 = nb * 64;
  int lane = threadIdx.x;
  int ncol = lane & 15;
  int kb8  = (lane >> 4) * 8;
  int kb16 = (lane >> 4) * 16;
  const half_t* arow0 = wp + (size_t)(o0 + ncol) * C_;
  const half_t* arow1 = wp + (size_t)(o0 + 16 + ncol) * C_;
  const half_t* bbase = aT + ((size_t)b * T_ + t0 + ncol) * C_;
  v8f acc[2][4] = {};
#pragma unroll 2
  for (int k0 = 0; k0 < C_; k0 += 32) {
    v16h A0 = load_a_frag(arow0 + k0, kb8);
    v16h A1 = load_a_frag(arow1 + k0, kb8);
#pragma unroll
    for (int j = 0; j < 4; ++j) {
      v16h Bf = *(const v16h*)(bbase + (size_t)j * 16 * C_ + k0 + kb16);
      acc[0][j] = wmma16(A0, Bf, acc[0][j]);
      acc[1][j] = wmma16(A1, Bf, acc[1][j]);
    }
  }
#pragma unroll
  for (int i = 0; i < 2; ++i) {
#pragma unroll
    for (int j = 0; j < 4; ++j) {
#pragma unroll
      for (int r = 0; r < 8; ++r) {
        int m  = r + ((lane >> 4) << 3);
        int oc = o0 + i * 16 + m;
        int t  = t0 + j * 16 + ncol;
        size_t o_ = ((size_t)b * C_ + oc) * T_ + t;
        out[o_] = x[o_] + acc[i][j][r] + bias[oc];
      }
    }
  }
}

// ---------------------------------------------------------------------------
extern "C" void kernel_launch(void* const* d_in, const int* in_sizes, int n_in,
                              void* d_out, int out_size, void* d_ws, size_t ws_size,
                              hipStream_t stream) {
  (void)in_sizes; (void)n_in; (void)out_size; (void)ws_size;
  const float* x     = (const float*)d_in[0];
  const float* cin   = (const float*)d_in[1];
  const float* gamma = (const float*)d_in[2];
  const float* beta  = (const float*)d_in[3];
  const float* w_qkv = (const float*)d_in[4];
  const float* b_qkv = (const float*)d_in[5];
  const float* w_c   = (const float*)d_in[6];
  const float* b_c   = (const float*)d_in[7];
  const float* w_p   = (const float*)d_in[8];
  const float* b_p   = (const float*)d_in[9];
  float* out = (float*)d_out;

  char* ws = (char*)d_ws;
  size_t off = 0;
  auto alloc = [&](size_t bytes) -> void* {
    void* p = ws + off;
    off += (bytes + 255) & ~(size_t)255;
    return p;
  };
  half_t* wqkv_h = (half_t*)alloc((size_t)3 * C_ * C_ * 2);
  half_t* wc_h   = (half_t*)alloc((size_t)2 * C_ * CROSS_ * 2);
  half_t* wp_h   = (half_t*)alloc((size_t)C_ * C_ * 2);
  half_t* xn_t   = (half_t*)alloc((size_t)B_ * T_ * C_ * 2);
  half_t* qT     = (half_t*)alloc((size_t)B_ * HEADS_ * T_ * HC_ * 2);
  half_t* kT     = (half_t*)alloc((size_t)B_ * HEADS_ * S_PAD * HC_ * 2);
  half_t* vx     = (half_t*)alloc((size_t)B_ * HEADS_ * HC_ * S_PAD * 2);
  half_t* aT     = (half_t*)alloc((size_t)B_ * T_ * C_ * 2);
  // total ~90.7 MB of workspace (fits in the 192 MB L2)

  cvt_kernel<<<(3 * C_ * C_ + 255) / 256, 256, 0, stream>>>(
      w_qkv, w_c, w_p, wqkv_h, wc_h, wp_h);
  gn_kernel<<<B_ * 32, 256, 0, stream>>>(x, gamma, beta, xn_t);
  qkv_kernel<<<B_ * 48 * 16, 32, 0, stream>>>(wqkv_h, b_qkv, xn_t, qT, kT, vx);
  ckv_kernel<<<B_ * 64 * 5, 32, 0, stream>>>(wc_h, b_c, cin, kT, vx);
  attn_kernel<<<B_ * HEADS_ * (T_ / 32), 32, 0, stream>>>(qT, kT, vx, aT);
  proj_kernel<<<B_ * 16 * 16, 32, 0, stream>>>(wp_h, b_p, aT, x, out);
}
